// MultiHeadCrossAttention_89137751261458
// MI455X (gfx1250) — compile-verified, matrix operand order checked by reading, not run on hardware
//
#include <hip/hip_runtime.h>
#include <math.h>

#define DIM 1152
#define NHEADS 16
#define HDIM 72
#define HPAD 96      // padded head dim for Q/K (3 x K=32 chunks)
#define VPAD 80      // padded head dim for V (5 x 16 output tiles)
#define Bsz 4
#define Nq 4096
#define Mkv 512

typedef __bf16 v16bf __attribute__((ext_vector_type(16)));
typedef float  v8f   __attribute__((ext_vector_type(8)));

__device__ __forceinline__ v8f wmma_bf16(v16bf a, v16bf b, v8f c) {
  // D = A(16x32) * B(32x16) + C, fp32 accumulate
  return __builtin_amdgcn_wmma_f32_16x16x32_bf16(false, a, false, b, (short)0, c,
                                                 false, false);
}

// ---------------------------------------------------------------------------
// Elementwise fp32 -> bf16 pack (n divisible by 4).
// ---------------------------------------------------------------------------
__global__ __launch_bounds__(256)
void cvt_f32_bf16(const float* __restrict__ src, __bf16* __restrict__ dst,
                  long n) {
  long i = ((long)blockIdx.x * 256 + threadIdx.x) * 4;
  if (i < n) {
    float4 v = *(const float4*)(src + i);
    dst[i + 0] = (__bf16)v.x;
    dst[i + 1] = (__bf16)v.y;
    dst[i + 2] = (__bf16)v.z;
    dst[i + 3] = (__bf16)v.w;
  }
}

// ---------------------------------------------------------------------------
// Weight transpose + convert: W (K x Nc) fp32 -> Wt (Nc x K) bf16.
// 32x32 tiles through LDS; grid = (Nc/32, K/32), 256 threads.
// ---------------------------------------------------------------------------
__global__ __launch_bounds__(256)
void wt_pack(const float* __restrict__ W, __bf16* __restrict__ Wt, int K,
             int Nc) {
  __shared__ __bf16 tile[32][33];
  const int kb = blockIdx.y * 32, nb = blockIdx.x * 32;
  const int tx = threadIdx.x & 31, ty = threadIdx.x >> 5;  // ty: 0..7
#pragma unroll
  for (int r = ty; r < 32; r += 8)
    tile[r][tx] = (__bf16)W[(size_t)(kb + r) * Nc + nb + tx];
  __syncthreads();
#pragma unroll
  for (int r = ty; r < 32; r += 8)
    Wt[(size_t)(nb + r) * K + kb + tx] = tile[tx][r];
}

// ---------------------------------------------------------------------------
// LDS-free GEMM: C(rows x Nc) = A(rows x K) * Wt(Nc x K)^T + bias, fp32 out.
// A, Wt bf16 row-major; both operands K-contiguous -> WMMA fragments load
// directly from global (32B per lane). Block 256x128, 8 waves, 64x64/wave
// (4x4 WMMA tiles): 16 WMMAs per 8 b128-pair loads per K-chunk.
// rows % 256 == 0, Nc % 128 == 0, K % 32 == 0.
// ---------------------------------------------------------------------------
__global__ __launch_bounds__(256)
void gemm_direct(const __bf16* __restrict__ A, const __bf16* __restrict__ Wt,
                 const float* __restrict__ bias, float* __restrict__ C, int K,
                 int Nc) {
  const int tid  = threadIdx.x;
  const int wave = tid >> 5;
  const int lane = tid & 31;
  const int row0 = blockIdx.y * 256;
  const int col0 = blockIdx.x * 128;
  const int rw   = (wave & 3) * 64;   // wave row offset
  const int cw   = (wave >> 2) * 64;  // wave col offset
  const int lrow = lane & 15;
  const int lkk  = (lane >> 4) * 16;

  const __bf16* apt = A + (size_t)(row0 + rw + lrow) * K + lkk;
  const __bf16* bpt = Wt + (size_t)(col0 + cw + lrow) * K + lkk;

  v8f acc[4][4];
#pragma unroll
  for (int i = 0; i < 4; ++i)
#pragma unroll
    for (int j = 0; j < 4; ++j)
      acc[i][j] = (v8f){0.f, 0.f, 0.f, 0.f, 0.f, 0.f, 0.f, 0.f};

  for (int k0 = 0; k0 < K; k0 += 32) {
    v16bf af[4], bfr[4];
#pragma unroll
    for (int i = 0; i < 4; ++i)
      af[i] = *(const v16bf*)(apt + (size_t)(i * 16) * K + k0);
#pragma unroll
    for (int j = 0; j < 4; ++j)
      bfr[j] = *(const v16bf*)(bpt + (size_t)(j * 16) * K + k0);
#pragma unroll
    for (int i = 0; i < 4; ++i)
#pragma unroll
      for (int j = 0; j < 4; ++j)
        acc[i][j] = wmma_bf16(af[i], bfr[j], acc[i][j]);
  }

  // Epilogue: bias + fp32 store. C layout: reg r -> row r + 8*(lane/16),
  // col = lane%16 within each 16x16 tile.
  const int rsel = (lane >> 4) << 3;
#pragma unroll
  for (int i = 0; i < 4; ++i)
#pragma unroll
    for (int j = 0; j < 4; ++j) {
      int gcol = col0 + cw + j * 16 + lrow;
      float bv = bias[gcol];
#pragma unroll
      for (int r = 0; r < 8; ++r) {
        int grow = row0 + rw + i * 16 + r + rsel;
        C[(size_t)grow * Nc + gcol] = acc[i][j][r] + bv;
      }
    }
}

// ---------------------------------------------------------------------------
// Q pack: per-(b,n,h) RMSNorm over 72 elems, write bf16 (B,H,N,96), zero-pad.
// One wave per row; 8 rows per 256-thread block.
// ---------------------------------------------------------------------------
__global__ __launch_bounds__(256)
void q_pack(const float* __restrict__ qraw, const float* __restrict__ qnw,
            __bf16* __restrict__ qws) {
  const int wave = threadIdx.x >> 5, lane = threadIdx.x & 31;
  long row = (long)blockIdx.x * 8 + wave;   // over B*N*H
  int h = (int)(row % NHEADS);
  long t = row / NHEADS;
  int n = (int)(t % Nq);
  int b = (int)(t / Nq);

  const float* src = qraw + ((size_t)b * Nq + n) * DIM + h * HDIM;
  float v0 = src[lane];
  float v1 = src[lane + 32];
  float v2 = 0.f, w2 = 0.f;
  if (lane < 8) { v2 = src[lane + 64]; w2 = qnw[lane + 64]; }
  float ss = v0 * v0 + v1 * v1 + v2 * v2;
#pragma unroll
  for (int off = 16; off; off >>= 1) ss += __shfl_xor(ss, off, 32);
  float inv = rsqrtf(ss * (1.0f / HDIM) + 1e-6f);

  __bf16* dst = qws + (((size_t)(b * NHEADS + h) * Nq) + n) * HPAD;
  dst[lane]      = (__bf16)(v0 * inv * qnw[lane]);
  dst[lane + 32] = (__bf16)(v1 * inv * qnw[lane + 32]);
  dst[lane + 64] = (lane < 8) ? (__bf16)(v2 * inv * w2) : (__bf16)0.f;
}

// ---------------------------------------------------------------------------
// KV pack: per-(b,m,h): RMSNorm K -> (B,H,M,96) bf16; V -> transposed
// (B,H,80,512) bf16 (rows 72..79 zero) so P*V's B operand is K-contiguous.
// ---------------------------------------------------------------------------
__global__ __launch_bounds__(256)
void kv_pack(const float* __restrict__ kvraw, const float* __restrict__ knw,
             __bf16* __restrict__ kws, __bf16* __restrict__ vt) {
  const int wave = threadIdx.x >> 5, lane = threadIdx.x & 31;
  long row = (long)blockIdx.x * 8 + wave;   // over B*M*H
  int h = (int)(row % NHEADS);
  long t = row / NHEADS;
  int m = (int)(t % Mkv);
  int b = (int)(t / Mkv);
  size_t bh = (size_t)b * NHEADS + h;

  const float* ks = kvraw + ((size_t)b * Mkv + m) * (2 * DIM) + h * HDIM;
  const float* vs = ks + DIM;

  // ---- K: rmsnorm + pad to 96 ----
  float v0 = ks[lane];
  float v1 = ks[lane + 32];
  float v2 = 0.f, w2 = 0.f;
  if (lane < 8) { v2 = ks[lane + 64]; w2 = knw[lane + 64]; }
  float ss = v0 * v0 + v1 * v1 + v2 * v2;
#pragma unroll
  for (int off = 16; off; off >>= 1) ss += __shfl_xor(ss, off, 32);
  float inv = rsqrtf(ss * (1.0f / HDIM) + 1e-6f);

  __bf16* kd = kws + (bh * Mkv + m) * HPAD;
  kd[lane]      = (__bf16)(v0 * inv * knw[lane]);
  kd[lane + 32] = (__bf16)(v1 * inv * knw[lane + 32]);
  kd[lane + 64] = (lane < 8) ? (__bf16)(v2 * inv * w2) : (__bf16)0.f;

  // ---- V: transpose into (B,H,80,512), rows 72..79 zeroed ----
  __bf16* vtb = vt + (bh * VPAD) * Mkv + m;
  vtb[(size_t)lane * Mkv]        = (__bf16)vs[lane];
  vtb[(size_t)(lane + 32) * Mkv] = (__bf16)vs[lane + 32];
  if (lane < 16) {
    int d = lane + 64;                 // 64..79
    float v = 0.f;
    if (d < HDIM) v = vs[d];
    vtb[(size_t)d * Mkv] = (__bf16)v;
  }
}

// ---------------------------------------------------------------------------
// Attention: one 128-thread WG per (b, h, 16-row Q tile).
// Phase 1: scores (16 x 512) via bf16 WMMA (d padded to 96), scale+mask, LDS.
// Phase 2: cooperative softmax, probs stored bf16 in LDS (A-fragment layout).
// Phase 3: P(16x512) * V(512x80) via WMMA, store bf16 (d<72) to ao.
// ---------------------------------------------------------------------------
__global__ __launch_bounds__(128)
void attn(const __bf16* __restrict__ qws, const __bf16* __restrict__ kws,
          const __bf16* __restrict__ vtws, const int* __restrict__ kv_seqlen,
          __bf16* __restrict__ ao) {
  __shared__ float  sc[16][Mkv];        // 32 KB fp32 scores
  __shared__ __bf16 pb[16][Mkv];        // 16 KB bf16 probs
  __shared__ float  red[16][8];
  __shared__ float  rmaxs[16], rsums[16];

  const int NT = Nq / 16;               // 256 tiles along N
  int gid = blockIdx.x;
  int nt = gid & (NT - 1);
  int h  = (gid >> 8) & (NHEADS - 1);
  int b  = gid >> 12;
  int n0 = nt * 16;

  const int tid = threadIdx.x, wave = tid >> 5, lane = tid & 31;
  const int lrow = lane & 15;
  const int lkk  = (lane >> 4) * 16;
  const int rsel = (lane >> 4) << 3;
  const int sl = kv_seqlen[b];
  const float scale = 0.11785113019775793f;  // 1/sqrt(72)

  const size_t bh = (size_t)b * NHEADS + h;
  const __bf16* qbase = qws + (bh * Nq + n0) * HPAD;
  const __bf16* kbase = kws + bh * Mkv * HPAD;
  const __bf16* vbase = vtws + bh * VPAD * Mkv;

  // Q A-fragments: lane = row n0+lrow, 3 chunks of K=32 over padded d=96
  v16bf qf[3];
  {
    const __bf16* p = qbase + (size_t)lrow * HPAD + lkk;
    qf[0] = *(const v16bf*)(p);
    qf[1] = *(const v16bf*)(p + 32);
    qf[2] = *(const v16bf*)(p + 64);
  }

  // ---- Phase 1: scores ----
  for (int mt = wave; mt < Mkv / 16; mt += 4) {
    const __bf16* kp = kbase + (size_t)(mt * 16 + lrow) * HPAD + lkk;
    v16bf k0 = *(const v16bf*)(kp);
    v16bf k1 = *(const v16bf*)(kp + 32);
    v16bf k2 = *(const v16bf*)(kp + 64);
    v8f c = (v8f){0.f, 0.f, 0.f, 0.f, 0.f, 0.f, 0.f, 0.f};
    c = wmma_bf16(qf[0], k0, c);
    c = wmma_bf16(qf[1], k1, c);
    c = wmma_bf16(qf[2], k2, c);
    int m = mt * 16 + lrow;
    bool valid = (m < sl);
#pragma unroll
    for (int r = 0; r < 8; ++r) {
      int nl = r + rsel;
      sc[nl][m] = valid ? (c[r] * scale) : -1e30f;
    }
  }
  __syncthreads();

  // ---- Phase 2: softmax (8 threads per row, 64 cols each) ----
  {
    const int r = tid >> 3, seg = tid & 7;
    const float* srow = &sc[r][seg * 64];
    float lm = -1e30f;
#pragma unroll 8
    for (int i = 0; i < 64; ++i) lm = fmaxf(lm, srow[i]);
    red[r][seg] = lm;
    __syncthreads();
    if (tid < 16) {
      float m0 = red[tid][0];
#pragma unroll
      for (int i = 1; i < 8; ++i) m0 = fmaxf(m0, red[tid][i]);
      rmaxs[tid] = m0;
    }
    __syncthreads();
    float rm = rmaxs[r];
    float lsum = 0.f;
#pragma unroll 8
    for (int i = 0; i < 64; ++i) lsum += __expf(srow[i] - rm);
    red[r][seg] = lsum;
    __syncthreads();
    if (tid < 16) {
      float s0 = 0.f;
#pragma unroll
      for (int i = 0; i < 8; ++i) s0 += red[tid][i];
      rsums[tid] = s0;
    }
    __syncthreads();
    float rinv = 1.0f / rsums[r];
    __bf16* prow = &pb[r][seg * 64];
#pragma unroll 8
    for (int i = 0; i < 64; ++i)
      prow[i] = (__bf16)(__expf(srow[i] - rm) * rinv);
  }
  __syncthreads();

  // ---- Phase 3: out = P * V, d padded to 80 (5 tiles), store d<72 ----
  for (int dt = wave; dt < VPAD / 16; dt += 4) {
    v8f c = (v8f){0.f, 0.f, 0.f, 0.f, 0.f, 0.f, 0.f, 0.f};
    const __bf16* vp0 = vbase + (size_t)(dt * 16 + lrow) * Mkv + lkk;
#pragma unroll
    for (int mc = 0; mc < Mkv / 32; ++mc) {
      v16bf af = *(const v16bf*)&pb[lrow][mc * 32 + lkk];
      v16bf bfr = *(const v16bf*)(vp0 + mc * 32);
      c = wmma_bf16(af, bfr, c);
    }
    int d = dt * 16 + lrow;
    if (d < HDIM) {
#pragma unroll
      for (int r = 0; r < 8; ++r) {
        int n = n0 + r + rsel;
        ao[((size_t)b * Nq + n) * DIM + h * HDIM + d] = (__bf16)c[r];
      }
    }
  }
}

// ---------------------------------------------------------------------------
extern "C" void kernel_launch(void* const* d_in, const int* in_sizes, int n_in,
                              void* d_out, int out_size, void* d_ws,
                              size_t ws_size, hipStream_t stream) {
  const float* x      = (const float*)d_in[0];
  const float* cond   = (const float*)d_in[1];
  const int*   kvlen  = (const int*)d_in[2];
  const float* q_w    = (const float*)d_in[3];
  const float* q_b    = (const float*)d_in[4];
  const float* kv_w   = (const float*)d_in[5];
  const float* kv_b   = (const float*)d_in[6];
  const float* proj_w = (const float*)d_in[7];
  const float* proj_b = (const float*)d_in[8];
  const float* qn_w   = (const float*)d_in[9];
  const float* kn_w   = (const float*)d_in[10];
  float* out = (float*)d_out;

  // Workspace layout. ao aliases xb (xb dead after GEMM-1, ao written later).
  char* ws = (char*)d_ws;
  __bf16* xb     = (__bf16*)(ws);                    // 37,748,736 B
  __bf16* ao     = (__bf16*)(ws);                    // alias of xb
  __bf16* cb     = (__bf16*)(ws + 37748736ull);      //  4,718,592 B
  __bf16* qwt    = (__bf16*)(ws + 42467328ull);      //  2,654,208 B
  __bf16* kvwt   = (__bf16*)(ws + 45121536ull);      //  5,308,416 B
  __bf16* pwt    = (__bf16*)(ws + 50429952ull);      //  2,654,208 B
  float*  q_raw  = (float*)(ws + 53084160ull);       // 75,497,472 B
  float*  kv_raw = (float*)(ws + 128581632ull);      // 18,874,368 B
  __bf16* qws    = (__bf16*)(ws + 147456000ull);     // 50,331,648 B
  __bf16* kws    = (__bf16*)(ws + 197787648ull);     //  6,291,456 B
  __bf16* vtws   = (__bf16*)(ws + 204079104ull);     //  5,242,880 B

  const long nx = (long)Bsz * Nq * DIM;              // 18,874,368
  const long nc = (long)Bsz * Mkv * DIM;             //  2,359,296

  // 0) Pack activations to bf16; pack weights transposed (Nc x K) bf16.
  cvt_f32_bf16<<<(int)(nx / 1024), 256, 0, stream>>>(x, xb, nx);
  cvt_f32_bf16<<<(int)(nc / 1024), 256, 0, stream>>>(cond, cb, nc);
  wt_pack<<<dim3(DIM / 32, DIM / 32), 256, 0, stream>>>(q_w, qwt, DIM, DIM);
  wt_pack<<<dim3((2 * DIM) / 32, DIM / 32), 256, 0, stream>>>(kv_w, kvwt, DIM,
                                                              2 * DIM);
  wt_pack<<<dim3(DIM / 32, DIM / 32), 256, 0, stream>>>(proj_w, pwt, DIM, DIM);

  // 1) Projections (LDS-free bf16 WMMA GEMMs, fp32 accumulate + bias)
  gemm_direct<<<dim3(DIM / 128, (Bsz * Nq) / 256), 256, 0, stream>>>(
      xb, qwt, q_b, q_raw, DIM, DIM);
  gemm_direct<<<dim3((2 * DIM) / 128, (Bsz * Mkv) / 256), 256, 0, stream>>>(
      cb, kvwt, kv_b, kv_raw, DIM, 2 * DIM);

  // 2) RMSNorm + bf16 repack (Q/K padded to 96, V transposed + padded to 80)
  q_pack<<<(Bsz * Nq * NHEADS) / 8, 256, 0, stream>>>(q_raw, qn_w, qws);
  kv_pack<<<(Bsz * Mkv * NHEADS) / 8, 256, 0, stream>>>(kv_raw, kn_w, kws,
                                                        vtws);

  // 3) Attention (one WG per (b, h, 16-row tile))
  attn<<<Bsz * NHEADS * (Nq / 16), 128, 0, stream>>>(qws, kws, vtws, kvlen,
                                                     ao);

  // 4) Output projection -> fp32 d_out
  gemm_direct<<<dim3(DIM / 128, (Bsz * Nq) / 256), 256, 0, stream>>>(
      ao, pwt, proj_b, out, DIM, DIM);
}
